// RegionProposalNetwork_82351702933519
// MI455X (gfx1250) — matrix operand order repaired and checked by reading, not compiled, not run on hardware
//
#include <hip/hip_runtime.h>
#include <hip/hip_bf16.h>
#include <stdint.h>

// ---------------------------------------------------------------------------
// CDNA5 (gfx1250) RPN: 3x3 conv + 1x1 heads as bf16 WMMA GEMMs, then
// proposal generation + greedy NMS. wave32, V_WMMA_F32_16X16X32_BF16.
// Conv uses 32(M)x64(N) register tile per wave: 8 wmma per K-step against
// 12 b128 loads (21 FLOP/byte of fragment traffic vs 8 for naive 16x16).
// ---------------------------------------------------------------------------

typedef __attribute__((ext_vector_type(16))) __bf16 v16bf;
typedef __attribute__((ext_vector_type(8)))  __bf16 v8bf;
typedef __attribute__((ext_vector_type(8)))  float  v8f;

#define NIMG   4
#define HH     50
#define WW     50
#define CIN    512
#define COUT   512
#define NPOS   2500          // 50*50
#define MT     157           // ceil(2500/16)  m-tiles (heads)
#define MT32   79            // ceil(2500/32)  m-tiles (conv)
#define KSTEPS1 144          // 512*9/32
#define KSTEPS2 16           // 512/32
#define NANCH  22500         // 2500*9
#define NPOST  2000

// output offsets (floats)
#define OFF_LOCS   0
#define OFF_SCORES 360000
#define OFF_ROIS   540000
#define OFF_RIDX   572000
#define OFF_ANCH   580000

__device__ __forceinline__ unsigned short f2bf(float f) {
  union { float f; unsigned u; } v; v.f = f;
  unsigned u = v.u;
  unsigned r = u + 0x7FFFu + ((u >> 16) & 1u);   // round-to-nearest-even
  return (unsigned short)(r >> 16);
}

__device__ __forceinline__ v16bf ldfrag(const __bf16* p, int stridePairs) {
  // two 16B halves: [p, p+8) and [p+stridePairs*8, ...)
  const v8bf* q = (const v8bf*)p;
  v8bf lo = q[0];
  v8bf hi = q[stridePairs];
  return __builtin_shufflevector(lo, hi, 0, 1, 2, 3, 4, 5, 6, 7,
                                 8, 9, 10, 11, 12, 13, 14, 15);
}

// ---- Stage 0: x (NCHW f32) -> zero-padded NHWC bf16 [4][52][52][512] ------
__global__ void k_pad_x(const float* __restrict__ x, unsigned short* __restrict__ xpad) {
  long idx = (long)blockIdx.x * blockDim.x + threadIdx.x;
  const long N = (long)NIMG * 52 * 52 * CIN;
  if (idx >= N) return;
  int ci = idx & 511;
  long t = idx >> 9;
  int ix = (int)(t % 52); t /= 52;
  int iy = (int)(t % 52); t /= 52;
  int b  = (int)t;
  float v = 0.f;
  int oy = iy - 1, ox = ix - 1;
  if (oy >= 0 && oy < HH && ox >= 0 && ox < WW)
    v = x[(((long)b * CIN + ci) * HH + oy) * WW + ox];
  xpad[idx] = f2bf(v);
}

// ---- Stage 1: pack conv1_w (OIHW) into B-fragment layout ------------------
// wp1[cot(32)][kstep(144)][lane(32)][e(16)], K order: k = tap*512 + ci
// B fragment (16-bit, 32x16): lanes 0-15 hold K=0..7,16..23, lanes 16-31 K=8..15,24..31.
__global__ void k_pack_w1(const float* __restrict__ w, unsigned short* __restrict__ wp) {
  int idx = blockIdx.x * blockDim.x + threadIdx.x;   // 32*144*32*16 = 2359296
  if (idx >= 32 * KSTEPS1 * 32 * 16) return;
  int e     = idx & 15;
  int lane  = (idx >> 4) & 31;
  int kstep = (idx >> 9) % KSTEPS1;
  int cot   = (idx >> 9) / KSTEPS1;
  int hi    = (lane >> 4) & 1;
  int k     = kstep * 32 + hi * 8 + (e & 7) + (e >> 3) * 16;
  int tap   = k >> 9;
  int ci    = k & 511;
  int n     = cot * 16 + (lane & 15);
  wp[idx] = f2bf(w[((long)n * CIN + ci) * 9 + tap]);
}

// ---- Stage 2: pack loc_w/score_w into B-fragment layout -------------------
// wp2[cot2(4)][kstep(16)][lane(32)][e(16)], n<36: loc, 36..53: score, else 0
__global__ void k_pack_w2(const float* __restrict__ lw, const float* __restrict__ sw,
                          unsigned short* __restrict__ wp) {
  int idx = blockIdx.x * blockDim.x + threadIdx.x;   // 4*16*32*16 = 32768
  if (idx >= 4 * KSTEPS2 * 32 * 16) return;
  int e     = idx & 15;
  int lane  = (idx >> 4) & 31;
  int kstep = (idx >> 9) & 15;
  int cot   = idx >> 13;
  int hi    = (lane >> 4) & 1;
  int ci    = kstep * 32 + hi * 8 + (e & 7) + (e >> 3) * 16;
  int n     = cot * 16 + (lane & 15);
  float v = 0.f;
  if (n < 36)      v = lw[(long)n * CIN + ci];
  else if (n < 54) v = sw[(long)(n - 36) * CIN + ci];
  wp[idx] = f2bf(v);
}

// ---- Stage 3: 3x3 conv + bias + ReLU as WMMA implicit GEMM ----------------
// grid = NIMG*MT32 blocks, 256 threads (8 waves).
// Wave w computes rows [mtile*32, +32) x couts [w*64, +64): 8 accumulators.
__global__ __launch_bounds__(256) void k_conv3x3(
    const unsigned short* __restrict__ xpad, const unsigned short* __restrict__ wp,
    const float* __restrict__ bias, unsigned short* __restrict__ h) {
  int lane  = threadIdx.x & 31;
  int wave  = threadIdx.x >> 5;               // 0..7
  int b     = blockIdx.x / MT32;
  int mtile = blockIdx.x % MT32;
  int hi    = (lane >> 4) & 1;

  int r0 = mtile * 32 + (lane & 15);
  int r1 = r0 + 16;
  int m0 = r0 < NPOS ? r0 : NPOS - 1;         // clamp tail rows (store-masked)
  int m1 = r1 < NPOS ? r1 : NPOS - 1;
  int oy0 = m0 / WW, ox0 = m0 % WW;
  int oy1 = m1 / WW, ox1 = m1 % WW;

  const __bf16* xb = (const __bf16*)xpad;
  // B fragment base for this wave's first cot tile (cot = wave*4 + ni)
  const __bf16* wb = (const __bf16*)wp + (((long)(wave * 4) * KSTEPS1) * 32 + lane) * 16;
  const long cotStride = (long)KSTEPS1 * 32 * 16;   // halfs between cot tiles

  v8f acc[8] = {};                            // [mi*4 + ni]
  for (int tap = 0; tap < 9; ++tap) {
    int ry = tap / 3, rx = tap % 3;
    const __bf16* rp0 = xb + ((((long)b * 52 + (oy0 + ry)) * 52 + (ox0 + rx)) << 9) + hi * 8;
    const __bf16* rp1 = xb + ((((long)b * 52 + (oy1 + ry)) * 52 + (ox1 + rx)) << 9) + hi * 8;
    const __bf16* wtap = wb + (long)tap * 16 * 512;
#pragma unroll 2
    for (int ks = 0; ks < 16; ++ks) {
      v16bf a0 = ldfrag(rp0 + ks * 32, 2);    // +16 halfs
      v16bf a1 = ldfrag(rp1 + ks * 32, 2);
      const __bf16* wks = wtap + (long)ks * 512;
#pragma unroll
      for (int ni = 0; ni < 4; ++ni) {
        v16bf bf = ldfrag(wks + ni * cotStride, 1);   // 32B contiguous
        acc[ni]     = __builtin_amdgcn_wmma_f32_16x16x32_bf16(false, a0, false, bf,
                                                              (short)0, acc[ni], false, false);
        acc[4 + ni] = __builtin_amdgcn_wmma_f32_16x16x32_bf16(false, a1, false, bf,
                                                              (short)0, acc[4 + ni], false, false);
      }
    }
  }

#pragma unroll
  for (int ni = 0; ni < 4; ++ni) {
    int n = wave * 64 + ni * 16 + (lane & 15);
    float bia = bias[n];
#pragma unroll
    for (int mi = 0; mi < 2; ++mi) {
#pragma unroll
      for (int r = 0; r < 8; ++r) {
        int mm = mtile * 32 + mi * 16 + r + hi * 8;
        if (mm < NPOS) {
          float v = acc[mi * 4 + ni][r] + bia;
          v = v > 0.f ? v : 0.f;
          h[(((long)b * NPOS + mm) << 9) + n] = f2bf(v);
        }
      }
    }
  }
}

// ---- Stage 4: 1x1 heads (loc 36 + score 18) as WMMA GEMM ------------------
// grid = NIMG*MT, block = 128 (4 waves), wave -> head-channel tile of 16.
__global__ __launch_bounds__(128) void k_heads(
    const unsigned short* __restrict__ h, const unsigned short* __restrict__ wp2,
    const float* __restrict__ lb, const float* __restrict__ sb, float* __restrict__ out) {
  int lane  = threadIdx.x & 31;
  int cot   = threadIdx.x >> 5;               // 0..3
  int b     = blockIdx.x / MT;
  int mtile = blockIdx.x % MT;
  int hi    = (lane >> 4) & 1;
  int mrow  = mtile * 16 + (lane & 15);
  int m     = mrow < NPOS ? mrow : NPOS - 1;

  const __bf16* hb = (const __bf16*)h + (((long)b * NPOS + m) << 9) + hi * 8;
  const __bf16* wb = (const __bf16*)wp2 + (((long)cot * KSTEPS2) * 32 + lane) * 16;

  v8f acc = {};
#pragma unroll 4
  for (int ks = 0; ks < KSTEPS2; ++ks) {
    v16bf af = ldfrag(hb + ks * 32, 2);
    v16bf bf = ldfrag(wb + (long)ks * 512, 1);
    acc = __builtin_amdgcn_wmma_f32_16x16x32_bf16(false, af, false, bf,
                                                  (short)0, acc, false, false);
  }

  int n = cot * 16 + (lane & 15);
#pragma unroll
  for (int r = 0; r < 8; ++r) {
    int mm = mtile * 16 + r + hi * 8;
    if (mm >= NPOS) continue;
    long row = (long)b * NPOS + mm;
    float v = acc[r];
    if (n < 36)      out[OFF_LOCS + row * 36 + n] = v + lb[n];
    else if (n < 54) out[OFF_SCORES + row * 18 + (n - 36)] = v + sb[n - 36];
  }
}

// ---- Stage 5: anchors + loc2bbox + clip + min-size + fg score -------------
__global__ void k_props(const float* __restrict__ out_ro, const int* __restrict__ imh_p,
                        const int* __restrict__ imw_p, float* __restrict__ boxes,
                        float* __restrict__ scores, float* __restrict__ out_w) {
  int idx = blockIdx.x * blockDim.x + threadIdx.x;   // NIMG * 22500
  if (idx >= NIMG * NANCH) return;
  int a = idx % 9;
  int m = (idx / 9) % NPOS;
  int b = idx / NANCH;

  const float ratios[3] = {0.5f, 1.0f, 2.0f};
  const float scales[3] = {8.f, 16.f, 32.f};
  float r = ratios[a / 3], s = scales[a % 3];
  float ah = 16.f * s * __builtin_sqrtf(r);
  float aw = 16.f * s * __builtin_sqrtf(1.0f / r);
  int oy = m / WW, ox = m % WW;
  float sy = oy * 16.f, sx = ox * 16.f;
  float a0 = sy + 8.f - 0.5f * ah;
  float a1 = sx + 8.f - 0.5f * aw;
  float a2 = sy + 8.f + 0.5f * ah;
  float a3 = sx + 8.f + 0.5f * aw;
  if (b == 0) {
    float* ap = out_w + OFF_ANCH + (long)(m * 9 + a) * 4;
    ap[0] = a0; ap[1] = a1; ap[2] = a2; ap[3] = a3;
  }

  long row = (long)b * NPOS + m;
  const float* L = out_ro + OFF_LOCS + row * 36 + a * 4;
  float dy = L[0], dx = L[1], dh = L[2], dw = L[3];
  float hh_ = a2 - a0, ww_ = a3 - a1;
  float cy = a0 + 0.5f * hh_, cx = a1 + 0.5f * ww_;
  float ncy = dy * hh_ + cy, ncx = dx * ww_ + cx;
  float nh = __expf(dh) * hh_, nw = __expf(dw) * ww_;
  float imh = (float)imh_p[0], imw = (float)imw_p[0];
  float y1 = fminf(fmaxf(ncy - 0.5f * nh, 0.f), imh);
  float x1 = fminf(fmaxf(ncx - 0.5f * nw, 0.f), imw);
  float y2 = fminf(fmaxf(ncy + 0.5f * nh, 0.f), imh);
  float x2 = fminf(fmaxf(ncx + 0.5f * nw, 0.f), imw);

  const float* S = out_ro + OFF_SCORES + row * 18 + a * 2;
  float fg = 1.f / (1.f + __expf(S[0] - S[1]));   // softmax pair -> sigmoid
  if ((y2 - y1) < 16.f || (x2 - x1) < 16.f) fg = -__builtin_inff();

  long gi = (long)b * NANCH + m * 9 + a;
  boxes[gi * 4 + 0] = y1;
  boxes[gi * 4 + 1] = x1;
  boxes[gi * 4 + 2] = y2;
  boxes[gi * 4 + 3] = x2;
  scores[gi] = fg;
}

// ---- Stage 6: greedy NMS (argmax + parallel IoU suppression) --------------
__global__ __launch_bounds__(256) void k_nms(const float* __restrict__ boxes,
                                             float* __restrict__ scores,
                                             float* __restrict__ out) {
  int b = blockIdx.x;
  int tid = threadIdx.x;
  __shared__ float sv[256];
  __shared__ int   si[256];
  const float* bx = boxes + (long)b * NANCH * 4;
  float* sc   = scores + (long)b * NANCH;
  float* rois = out + OFF_ROIS + (long)b * NPOST * 4;
  float* ridx = out + OFF_RIDX + (long)b * NPOST;

  for (int t = 0; t < NPOST; ++t) {
    float best = -__builtin_inff(); int bi = -1;
    for (int i = tid; i < NANCH; i += 256) {
      float v = sc[i];
      if (v > best) { best = v; bi = i; }
    }
    sv[tid] = best; si[tid] = bi;
    __syncthreads();
    for (int off = 128; off > 0; off >>= 1) {
      if (tid < off && sv[tid + off] > sv[tid]) { sv[tid] = sv[tid + off]; si[tid] = si[tid + off]; }
      __syncthreads();
    }
    float bs = sv[0]; int idx = si[0];
    bool valid = (bs > -1e37f) && (idx >= 0);
    if (tid < 4) rois[t * 4 + tid] = valid ? bx[idx * 4 + tid] : 0.f;
    if (tid == 4) ridx[t] = (float)b;
    if (valid) {
      float by1 = bx[idx * 4 + 0], bx1 = bx[idx * 4 + 1];
      float by2 = bx[idx * 4 + 2], bx2 = bx[idx * 4 + 3];
      float barea = (by2 - by1) * (bx2 - bx1);
      for (int i = tid; i < NANCH; i += 256) {
        if (sc[i] <= -1e37f) continue;
        float y1 = fmaxf(by1, bx[i * 4 + 0]);
        float x1 = fmaxf(bx1, bx[i * 4 + 1]);
        float y2 = fminf(by2, bx[i * 4 + 2]);
        float x2 = fminf(bx2, bx[i * 4 + 3]);
        float inter = fmaxf(y2 - y1, 0.f) * fmaxf(x2 - x1, 0.f);
        float area = (bx[i * 4 + 2] - bx[i * 4 + 0]) * (bx[i * 4 + 3] - bx[i * 4 + 1]);
        float iou = inter / (area + barea - inter + 1e-9f);
        if (iou > 0.7f) sc[i] = -__builtin_inff();   // suppresses best itself too
      }
    }
    __syncthreads();
  }
}

// ---------------------------------------------------------------------------
extern "C" void kernel_launch(void* const* d_in, const int* in_sizes, int n_in,
                              void* d_out, int out_size, void* d_ws, size_t ws_size,
                              hipStream_t stream) {
  (void)in_sizes; (void)n_in; (void)out_size; (void)ws_size;
  const float* x   = (const float*)d_in[0];
  const float* w1  = (const float*)d_in[1];
  const float* b1  = (const float*)d_in[2];
  const float* lw  = (const float*)d_in[3];
  const float* lb  = (const float*)d_in[4];
  const float* sw  = (const float*)d_in[5];
  const float* sb  = (const float*)d_in[6];
  const int*   imh = (const int*)d_in[7];
  const int*   imw = (const int*)d_in[8];
  float* out = (float*)d_out;

  // workspace carve-up (256B aligned slabs)
  char* ws = (char*)d_ws;
  size_t o = 0;
  auto alloc = [&](size_t bytes) { char* p = ws + o; o += (bytes + 255) & ~(size_t)255; return p; };
  unsigned short* xpad = (unsigned short*)alloc((size_t)NIMG * 52 * 52 * CIN * 2);      // 11.1 MB
  unsigned short* wp1  = (unsigned short*)alloc((size_t)32 * KSTEPS1 * 32 * 16 * 2);    // 4.7 MB
  unsigned short* hbuf = (unsigned short*)alloc((size_t)NIMG * NPOS * COUT * 2);        // 10.2 MB
  unsigned short* wp2  = (unsigned short*)alloc((size_t)4 * KSTEPS2 * 32 * 16 * 2);     // 64 KB
  float* boxes  = (float*)alloc((size_t)NIMG * NANCH * 4 * 4);                          // 1.44 MB
  float* scores = (float*)alloc((size_t)NIMG * NANCH * 4);                              // 0.36 MB

  {
    long n = (long)NIMG * 52 * 52 * CIN;
    k_pad_x<<<(unsigned)((n + 255) / 256), 256, 0, stream>>>(x, xpad);
  }
  k_pack_w1<<<(32 * KSTEPS1 * 32 * 16) / 256, 256, 0, stream>>>(w1, wp1);
  k_pack_w2<<<(4 * KSTEPS2 * 32 * 16) / 256, 256, 0, stream>>>(lw, sw, wp2);

  k_conv3x3<<<dim3(NIMG * MT32), 256, 0, stream>>>(xpad, wp1, b1, hbuf);
  k_heads<<<dim3(NIMG * MT), 128, 0, stream>>>(hbuf, wp2, lb, sb, out);

  k_props<<<(NIMG * NANCH + 255) / 256, 256, 0, stream>>>(out, imh, imw, boxes, scores, out);
  k_nms<<<NIMG, 256, 0, stream>>>(boxes, scores, out);
}
